// CirculantTransform_24223615549793
// MI455X (gfx1250) — matrix operand order repaired and checked by reading, not compiled, not run on hardware
//
#include <hip/hip_runtime.h>

// MI455X / gfx1250: wave32, WMMA f16 -> f32 accum.
// out[b] = circulant(seq_circ) @ input[b] @ circulant(hidden_circ)^T
// Two-stage tiled GEMM; circulant tiles gathered from generator vectors (never materialized).

static constexpr int SEQ = 4096;   // S
static constexpr int HID = 1024;   // H
static constexpr int NB  = 8;      // batch

typedef __attribute__((ext_vector_type(16))) _Float16 v16h;
typedef __attribute__((ext_vector_type(8)))  float    v8f;

union Frag32 {               // one 16x32 (A) or 32x16 (B) f16 WMMA operand = 8 VGPRs
    v16h         h;
    uint4        q[2];       // two 16-byte chunks
    unsigned int u[8];       // u[i] = VGPR i = (f16 K even | f16 K odd << 16)
    _Float16     e[16];
};

__device__ __forceinline__ unsigned short f16bits(float x) {
    union { _Float16 f; unsigned short s; } cv; cv.f = (_Float16)x; return cv.s;
}
__device__ __forceinline__ unsigned int pack2(_Float16 lo, _Float16 hi) {
    union { _Float16 f; unsigned short s; } a, b; a.f = lo; b.f = hi;
    return (unsigned)a.s | ((unsigned)b.s << 16);
}
__device__ __forceinline__ int swz(int c) {        // 16B-chunk XOR swizzle: 2-way banks
    return c ^ (((c >> 4) & 3) << 2);
}

// ---------------------------------------------------------------------------
// Stage 1:  X[s,h] = sum_t seq[(s-t) mod S] * E[t,h]   (per batch), X stored f16
// Block: 256 thr = 8 waves; tile 256(s) x 128(h): each wave owns 32 s-rows
// (two A-fragments sharing every B-fragment). K = S in steps of 32.
// Circulant A-frag: pairs (seq[c-k], seq[c-k-1]) ascend in rseq[t]=seq[-t mod S];
// parity-duplicated u32 pair table -> 8 aligned ds_load_b32 per fragment.
// ---------------------------------------------------------------------------
__global__ __launch_bounds__(256)
void circ_stage1(const float* __restrict__ inp, const float* __restrict__ seqc,
                 _Float16* __restrict__ Xb, int b)
{
    __shared__ __align__(16) unsigned int spair[2][SEQ / 2 + 16]; // +16 dw: de-alias parity banks
    __shared__ __align__(16) _Float16 etile[2][128][40];          // [h][k], 80B rows, dbl-buffered

    const int tid  = threadIdx.x;
    const int lane = tid & 31;
    const int wave = tid >> 5;
    const int m    = lane & 15;             // matrix row within 16 (both lane halves)
    const int hf   = lane >> 4;             // lane half selects K sub-block
    const int s0   = blockIdx.y * 256;
    const int h0   = blockIdx.x * 128;

    // spair[q][p] = (rseq[2p+q], rseq[2p+q+1]) packed, rseq[t] = f16(seq[(-t) mod S])
    for (int p = tid; p < SEQ / 2; p += 256) {
        const _Float16 a0 = (_Float16)seqc[(SEQ - 2*p    ) & (SEQ - 1)];
        const _Float16 a1 = (_Float16)seqc[(SEQ - 2*p - 1) & (SEQ - 1)];
        const _Float16 a2 = (_Float16)seqc[(SEQ - 2*p - 2) & (SEQ - 1)];
        spair[0][p] = pack2(a0, a1);
        spair[1][p] = pack2(a1, a2);
    }

    const v8f vzero = {0.f,0.f,0.f,0.f,0.f,0.f,0.f,0.f};
    v8f acc0[8], acc1[8];
#pragma unroll
    for (int j = 0; j < 8; ++j) { acc0[j] = vzero; acc1[j] = vzero; }

    const int srow = s0 + wave * 32 + m;                  // fragment 0; fragment 1 = +16
    const unsigned int* __restrict__ P = spair[srow & 1]; // parity invariant (16, 32 even)

    // E-slab staging: thread owns column h = tid&127, k-range (tid>>7)*16..+15.
    // Global: 16 coalesced b32 column loads. LDS: 2 ds_store_b128 (2-way banks).
    const int sh = tid & 127;
    const int sk = (tid >> 7) * 16;
    float tr[16];
    auto tload = [&](int kk) {
        const float* p = inp + ((size_t)b * SEQ + kk * 32 + sk) * HID + h0 + sh;
#pragma unroll
        for (int i = 0; i < 16; ++i) tr[i] = p[(size_t)i * HID];
        if (kk + 1 < SEQ / 32)                            // gfx1250 global_prefetch_b8
            __builtin_prefetch(p + 32 * HID, 0, 0);
    };
    auto tstore = [&](int kk) {
        uint4 w0, w1;
        w0.x = pack2((_Float16)tr[ 0], (_Float16)tr[ 1]);
        w0.y = pack2((_Float16)tr[ 2], (_Float16)tr[ 3]);
        w0.z = pack2((_Float16)tr[ 4], (_Float16)tr[ 5]);
        w0.w = pack2((_Float16)tr[ 6], (_Float16)tr[ 7]);
        w1.x = pack2((_Float16)tr[ 8], (_Float16)tr[ 9]);
        w1.y = pack2((_Float16)tr[10], (_Float16)tr[11]);
        w1.z = pack2((_Float16)tr[12], (_Float16)tr[13]);
        w1.w = pack2((_Float16)tr[14], (_Float16)tr[15]);
        uint4* dst = (uint4*)&etile[kk & 1][sh][sk];      // 32B contiguous, 16B aligned
        dst[0] = w0;
        dst[1] = w1;
    };

    tload(0); tstore(0);

    for (int kk = 0; kk < SEQ / 32; ++kk) {
        __syncthreads();                      // prev-iteration stores of this buffer visible
        const bool more = (kk + 1 < SEQ / 32);
        if (more) tload(kk + 1);              // global loads issue early, hide under WMMAs

        const int k0 = kk * 32;
        // A frags (ISA 7.12.2): VGPR i holds K = kbase..kbase+1, kbase = 16*(i>=4)+8*hf+2*(i&3)
        Frag32 a0, a1;
#pragma unroll
        for (int i = 0; i < 8; ++i) {
            const int kb = ((i >> 2) << 4) + (hf << 3) + ((i & 3) << 1);
            const unsigned x0 = ((unsigned)(kb - srow + k0)) & (SEQ - 1);
            a0.u[i] = P[x0 >> 1];
            a1.u[i] = P[((x0 - 16u) & (SEQ - 1)) >> 1];   // rows +16
        }

        const _Float16 (*et)[40] = etile[kk & 1];
#pragma unroll
        for (int jj = 0; jj < 8; jj += 2) {   // B frag shared by both row-fragments
            Frag32 bf[2];
#pragma unroll
            for (int j = 0; j < 2; ++j) {
                const uint4* qp = (const uint4*)&et[16 * (jj + j) + m][hf * 8];
                bf[j].q[0] = qp[0];
                bf[j].q[1] = qp[2];
            }
#pragma unroll
            for (int j = 0; j < 2; ++j) {
                acc0[jj + j] = __builtin_amdgcn_wmma_f32_16x16x32_f16(
                    false, a0.h, false, bf[j].h, (short)0, acc0[jj + j], false, false);
                acc1[jj + j] = __builtin_amdgcn_wmma_f32_16x16x32_f16(
                    false, a1.h, false, bf[j].h, (short)0, acc1[jj + j], false, false);
            }
        }

        if (more) tstore(kk + 1);             // convert + LDS-store after compute
    }

    // D layout: VGPR r -> row = r + 8*hf, col = lane&15. Store f16 (matches stage-2 A loads).
#pragma unroll
    for (int j = 0; j < 8; ++j) {
        const int col = h0 + 16 * j + m;
#pragma unroll
        for (int r = 0; r < 8; ++r) {
            const int row = s0 + wave * 32 + r + hf * 8;
            Xb[(size_t)row * HID + col]        = (_Float16)acc0[j][r];
            Xb[(size_t)(row + 16) * HID + col] = (_Float16)acc1[j][r];
        }
    }
}

// ---------------------------------------------------------------------------
// Stage 2:  Out[s,h] = sum_g X[s,g] * hid[(h-g) mod H]
// Circulant is the B operand; its fragment depends only on (16j - 32kk) mod H
// -> 64 diagonal classes precomputed once into a 64 KB LDS table in WMMA-ready,
// XOR-swizzled form (2-way banked b128 reads). Each wave owns 32 s-rows.
// ---------------------------------------------------------------------------
__global__ __launch_bounds__(256)
void circ_stage2(const _Float16* __restrict__ Xb, const float* __restrict__ hidc,
                 float* __restrict__ out, int b)
{
    __shared__ __align__(16) unsigned int tbl[64 * 256];  // 64 cls x 64 swizzled 16B chunks

    const int tid  = threadIdx.x;
    const int lane = tid & 31;
    const int wave = tid >> 5;
    const int m    = lane & 15;
    const int hf   = lane >> 4;
    const int s0   = blockIdx.y * 256;
    const int h0   = blockIdx.x * 128;

    // Build class table: B[k][n] = hid[(h0 + 16*cls + n - k) mod H]
    // B layout: lane n = ln&15, VGPR i holds K = {16*(i>=4) + 8*(ln>=16) + 2*(i&3), +1}
    for (int idx = tid; idx < 64 * 32 * 8; idx += 256) {
        const int i  = idx & 7;
        const int ln = (idx >> 3) & 31;
        const int c  = idx >> 8;
        const int kb = ((i >> 2) << 4) + ((ln >> 4) << 3) + 2 * (i & 3);
        const unsigned base = (unsigned)(h0 + 16 * c + (ln & 15) - kb);
        const unsigned short lo = f16bits(hidc[ base       & (HID - 1)]);
        const unsigned short hi = f16bits(hidc[(base - 1u) & (HID - 1)]);
        const int chunk = ((i >> 2) << 5) + ln;           // halves 512B apart, then swizzled
        tbl[c * 256 + swz(chunk) * 4 + (i & 3)] = (unsigned)lo | ((unsigned)hi << 16);
    }
    __syncthreads();

    const v8f vzero = {0.f,0.f,0.f,0.f,0.f,0.f,0.f,0.f};
    v8f acc0[8], acc1[8];
#pragma unroll
    for (int j = 0; j < 8; ++j) { acc0[j] = vzero; acc1[j] = vzero; }

    const int q0off = swz(lane) * 4;                      // lane's two swizzled chunk offsets
    const int q1off = swz(32 + lane) * 4;

    const _Float16* xrow0 = Xb + (size_t)(s0 + wave * 32 + m) * HID;
    const _Float16* xrow1 = xrow0 + (size_t)16 * HID;
    auto aload = [&](int kk, const _Float16* xr, Frag32& f) {
        const _Float16* p = xr + kk * 32 + 8 * hf;        // two aligned 16B runs per lane
        f.q[0] = *(const uint4*)p;
        f.q[1] = *(const uint4*)(p + 16);
    };

    Frag32 a0, a1, an0, an1;
    aload(0, xrow0, an0); aload(0, xrow1, an1);

    for (int kk = 0; kk < HID / 32; ++kk) {
        a0 = an0; a1 = an1;
        if (kk + 1 < HID / 32) {                          // next A frags overlap WMMAs
            aload(kk + 1, xrow0, an0);
            aload(kk + 1, xrow1, an1);
        }
#pragma unroll
        for (int jj = 0; jj < 8; jj += 2) {               // B frag shared by both row-fragments
            Frag32 bf[2];
#pragma unroll
            for (int j = 0; j < 2; ++j) {
                const int cls = (jj + j - 2 * kk) & 63;
                const unsigned int* bp = &tbl[cls * 256];
                bf[j].q[0] = *(const uint4*)(bp + q0off);
                bf[j].q[1] = *(const uint4*)(bp + q1off);
            }
#pragma unroll
            for (int j = 0; j < 2; ++j) {
                acc0[jj + j] = __builtin_amdgcn_wmma_f32_16x16x32_f16(
                    false, a0.h, false, bf[j].h, (short)0, acc0[jj + j], false, false);
                acc1[jj + j] = __builtin_amdgcn_wmma_f32_16x16x32_f16(
                    false, a1.h, false, bf[j].h, (short)0, acc1[jj + j], false, false);
            }
        }
    }

    // Coalesced f32 stores (16 lanes contiguous per row)
#pragma unroll
    for (int j = 0; j < 8; ++j) {
        const int col = h0 + 16 * j + m;
#pragma unroll
        for (int r = 0; r < 8; ++r) {
            const size_t row = (size_t)b * SEQ + s0 + wave * 32 + r + hf * 8;
            out[row * HID + col]        = acc0[j][r];
            out[(row + 16) * HID + col] = acc1[j][r];
        }
    }
}

// ---------------------------------------------------------------------------
extern "C" void kernel_launch(void* const* d_in, const int* in_sizes, int n_in,
                              void* d_out, int out_size, void* d_ws, size_t ws_size,
                              hipStream_t stream)
{
    (void)in_sizes; (void)n_in; (void)out_size; (void)ws_size;
    const float* inp  = (const float*)d_in[0];   // [8,4096,1024] f32
    const float* seqc = (const float*)d_in[1];   // [4096] f32
    const float* hidc = (const float*)d_in[2];   // [1024] f32
    float*       out  = (float*)d_out;           // [8,4096,1024] f32
    _Float16*    Xb   = (_Float16*)d_ws;         // per-batch intermediate, S*H f16 = 8 MB

    dim3 grid(HID / 128, SEQ / 256);             // (8, 16)
    for (int b = 0; b < NB; ++b) {               // stream order serializes stage1(b)->stage2(b)
        circ_stage1<<<grid, 256, 0, stream>>>(inp, seqc, Xb, b);
        circ_stage2<<<grid, 256, 0, stream>>>(Xb, hidc, out, b);
    }
}
